// LSTMGraphPredictor_36739150250147
// MI455X (gfx1250) — matrix operand order repaired and checked by reading, not compiled, not run on hardware
//
#include <hip/hip_runtime.h>
#include <hip/hip_bf16.h>
#include <stdint.h>

// ---------------- problem constants ----------------
#define BATCH 256
#define SEQ   512
#define DIN   150
#define HID   1500
#define DPAD  160            // DIN padded to multiple of 32 (5 chunks)
#define HPAD  1536           // h row stride (f16), padded; only k<1504 consumed
#define KC_X  5              // 160/32 K-chunks for input projection
#define KC_H  47             // 1504/32 K-chunks for recurrent projection
#define NT    94             // ceil(HID/16) N tiles over hidden dim
#define MBT   8              // 256/32 M blocks (each wave does 2 x 16-row subtiles)
#define WAVES (NT * MBT)     // 752 waves per step
#define FRAG_HALFS 512       // one packed 32x16 B fragment = 32 lanes * 16 halfs

typedef _Float16 h16;
typedef __attribute__((ext_vector_type(8)))  _Float16 v8h;
typedef __attribute__((ext_vector_type(16))) _Float16 v16h;
typedef __attribute__((ext_vector_type(8)))  float    v8f;

// ---------------- helpers ----------------
__device__ __forceinline__ float sigmoidf_(float x) {
  return 1.0f / (1.0f + __expf(-x));
}
__device__ __forceinline__ float tanhf_(float x) {
  float e = __expf(2.0f * x);
  return 1.0f - 2.0f / (e + 1.0f);
}

// A fragment (16x32 f16), ISA layout: lane<16 -> K kb..kb+7 (elems 0-7) and
// kb+16..kb+23 (elems 8-15); lane>=16 uses kb+8. Two 16B loads per lane.
__device__ __forceinline__ v16h load_a_frag(const h16* __restrict__ p) {
  v8h lo = *(const v8h*)(p);
  v8h hi = *(const v8h*)(p + 16);
  return __builtin_shufflevector(lo, hi, 0, 1, 2, 3, 4, 5, 6, 7,
                                 8, 9, 10, 11, 12, 13, 14, 15);
}

// ---------------- prep kernels (one-time) ----------------
// x [B][S][D] f32 -> xh [S][B][DPAD] f16 (zero padded)
__global__ void pack_x_kernel(const float* __restrict__ x, h16* __restrict__ xh) {
  int idx = blockIdx.x * blockDim.x + threadIdx.x;
  int total = SEQ * BATCH * DPAD;
  if (idx >= total) return;
  int k = idx % DPAD;
  int b = (idx / DPAD) % BATCH;
  int s = idx / (DPAD * BATCH);
  float v = 0.0f;
  if (k < DIN) v = x[((size_t)b * SEQ + s) * DIN + k];
  xh[idx] = (h16)v;
}

// W [4H][K] f32 -> packed WMMA-B fragments.
// Flat fragment order: ((t*4 + g)*KC + c), each fragment 32 lanes * 16 halfs.
// Fragment element (lane, e) = W[g*HID + t*16 + (lane&15)][c*32 + (e<8 ? 8*(lane>>4)+e : 16+8*(lane>>4)+e-8)] (0 if OOB).
__global__ void pack_w_kernel(const float* __restrict__ W, h16* __restrict__ pack,
                              int K, int KC) {
  int idx = blockIdx.x * blockDim.x + threadIdx.x;    // (t,g,c,lane)
  int total = NT * 4 * KC * 32;
  if (idx >= total) return;
  int lane = idx & 31;
  int c = (idx >> 5) % KC;
  int g = ((idx >> 5) / KC) % 4;
  int t = (idx >> 5) / (KC * 4);
  int n = t * 16 + (lane & 15);
  v16h frag;
  #pragma unroll
  for (int e = 0; e < 16; ++e) {
    int k = c * 32 + ((e < 8) ? 8 * (lane >> 4) + e : 16 + 8 * (lane >> 4) + (e - 8));
    float v = 0.0f;
    if (n < HID && k < K) v = W[((size_t)(g * HID + n)) * K + k];
    frag[e] = (h16)v;
  }
  *(v16h*)(pack + (size_t)idx * 16) = frag;
}

__global__ void bias_kernel(const float* __restrict__ bi, const float* __restrict__ bh,
                            float* __restrict__ bias) {
  int i = blockIdx.x * blockDim.x + threadIdx.x;
  if (i < 4 * HID) bias[i] = bi[i] + bh[i];
}

__global__ void zero_kernel(uint32_t* __restrict__ p, int n) {
  int i = blockIdx.x * blockDim.x + threadIdx.x;
  if (i < n) p[i] = 0u;
}

// ---------------- recurrent step ----------------
// One wave: 32(M) x 16(N of H) tile, all 4 gates. K-phase 1 over x_t (KC_X
// chunks), phase 2 over h_in (KC_H chunks). Epilogue: LSTM cell update.
__launch_bounds__(128)
__global__ void lstm_step_kernel(const h16* __restrict__ xh,     // [S][B][DPAD]
                                 const h16* __restrict__ wip,    // packed W_ih
                                 const h16* __restrict__ whp,    // packed W_hh
                                 const float* __restrict__ bias, // [4H]
                                 const h16* __restrict__ h_in,   // [B][HPAD]
                                 h16* __restrict__ h_out,        // [B][HPAD]
                                 float* __restrict__ cst,        // [B][HID]
                                 float* __restrict__ hf32,       // [B][HID] (= d_out)
                                 int s) {
  const int lane = threadIdx.x & 31;
  const int wid = (blockIdx.x * blockDim.x + threadIdx.x) >> 5;
  if (wid >= WAVES) return;                       // wave-uniform (never taken: 188*4=752)
  const int nt = wid % NT;
  const int m0 = (wid / NT) * 32;

  v8f acc[2][4];
  #pragma unroll
  for (int u = 0; u < 2; ++u)
    #pragma unroll
    for (int g = 0; g < 4; ++g) acc[u][g] = (v8f){};

  const int mrow = lane & 15;
  const int koff = (lane < 16) ? 0 : 8;

  // -------- phase 1: x_t @ W_ih^T --------
  {
    const h16* a0 = xh + ((size_t)s * BATCH + (m0 + mrow)) * DPAD;
    const h16* a1 = xh + ((size_t)s * BATCH + (m0 + 16 + mrow)) * DPAD;
    const h16* bp = wip + (size_t)(nt * 4) * KC_X * FRAG_HALFS + lane * 16;
    #pragma unroll
    for (int c = 0; c < KC_X; ++c) {
      const int kb = c * 32 + koff;
      v16h af0 = load_a_frag(a0 + kb);
      v16h af1 = load_a_frag(a1 + kb);
      #pragma unroll
      for (int g = 0; g < 4; ++g) {
        v16h bf = *(const v16h*)(bp + (size_t)(g * KC_X + c) * FRAG_HALFS);
        acc[0][g] = __builtin_amdgcn_wmma_f32_16x16x32_f16(
            false, af0, false, bf, (short)0, acc[0][g], false, false);
        acc[1][g] = __builtin_amdgcn_wmma_f32_16x16x32_f16(
            false, af1, false, bf, (short)0, acc[1][g], false, false);
      }
    }
  }

  // -------- phase 2: h @ W_hh^T --------
  {
    const h16* a0 = h_in + (size_t)(m0 + mrow) * HPAD;
    const h16* a1 = h_in + (size_t)(m0 + 16 + mrow) * HPAD;
    const h16* bp = whp + (size_t)(nt * 4) * KC_H * FRAG_HALFS + lane * 16;
    #pragma unroll 4
    for (int c = 0; c < KC_H; ++c) {
      const int kb = c * 32 + koff;
      v16h af0 = load_a_frag(a0 + kb);
      v16h af1 = load_a_frag(a1 + kb);
      #pragma unroll
      for (int g = 0; g < 4; ++g) {
        v16h bf = *(const v16h*)(bp + (size_t)(g * KC_H + c) * FRAG_HALFS);
        acc[0][g] = __builtin_amdgcn_wmma_f32_16x16x32_f16(
            false, af0, false, bf, (short)0, acc[0][g], false, false);
        acc[1][g] = __builtin_amdgcn_wmma_f32_16x16x32_f16(
            false, af1, false, bf, (short)0, acc[1][g], false, false);
      }
    }
  }

  // -------- LSTM cell epilogue --------
  // C/D layout: elem r of v8f -> row m = r + 8*(lane>=16), col n = lane&15.
  const int n = nt * 16 + (lane & 15);
  if (n < HID) {
    const float bi_ = bias[n];
    const float bf_ = bias[HID + n];
    const float bg_ = bias[2 * HID + n];
    const float bo_ = bias[3 * HID + n];
    const int mr = (lane >= 16) ? 8 : 0;
    #pragma unroll
    for (int u = 0; u < 2; ++u) {
      #pragma unroll
      for (int r = 0; r < 8; ++r) {
        const int m = m0 + u * 16 + r + mr;
        const float gi = sigmoidf_(acc[u][0][r] + bi_);
        const float gf = sigmoidf_(acc[u][1][r] + bf_);
        const float gg = tanhf_(acc[u][2][r] + bg_);
        const float go = sigmoidf_(acc[u][3][r] + bo_);
        const size_t ci = (size_t)m * HID + n;
        const float cn = gf * cst[ci] + gi * gg;
        cst[ci] = cn;
        const float hn = go * tanhf_(cn);
        hf32[ci] = hn;                            // d_out: last step leaves h_final
        h_out[(size_t)m * HPAD + n] = (h16)hn;    // f16 operand for next step
      }
    }
  }
}

// ---------------- host side ----------------
extern "C" void kernel_launch(void* const* d_in, const int* in_sizes, int n_in,
                              void* d_out, int out_size, void* d_ws, size_t ws_size,
                              hipStream_t stream) {
  const float* x    = (const float*)d_in[0];
  const float* W_ih = (const float*)d_in[1];
  const float* W_hh = (const float*)d_in[2];
  const float* b_ih = (const float*)d_in[3];
  const float* b_hh = (const float*)d_in[4];
  float* out = (float*)d_out;

  // workspace carve-up (256B aligned)
  uint8_t* ws = (uint8_t*)d_ws;
  size_t off = 0;
  auto carve = [&](size_t bytes) -> void* {
    void* p = ws + off;
    off = (off + bytes + 255) & ~(size_t)255;
    return p;
  };
  h16*   xh     = (h16*)  carve((size_t)SEQ * BATCH * DPAD * 2);          // 41.9 MB
  h16*   wipack = (h16*)  carve((size_t)NT * 4 * KC_X * FRAG_HALFS * 2);  //  1.9 MB
  h16*   whpack = (h16*)  carve((size_t)NT * 4 * KC_H * FRAG_HALFS * 2);  // 18.1 MB
  h16*   hping  = (h16*)  carve((size_t)BATCH * HPAD * 2);                //  0.8 MB
  h16*   hpong  = (h16*)  carve((size_t)BATCH * HPAD * 2);                //  0.8 MB
  float* cbuf   = (float*)carve((size_t)BATCH * HID * 4);                 //  1.5 MB
  float* bias   = (float*)carve((size_t)4 * HID * 4);
  (void)ws_size; (void)in_sizes; (void)n_in; (void)out_size;

  // one-time prep
  {
    int total = SEQ * BATCH * DPAD;
    pack_x_kernel<<<(total + 255) / 256, 256, 0, stream>>>(x, xh);
  }
  {
    int total = NT * 4 * KC_X * 32;
    pack_w_kernel<<<(total + 255) / 256, 256, 0, stream>>>(W_ih, wipack, DIN, KC_X);
  }
  {
    int total = NT * 4 * KC_H * 32;
    pack_w_kernel<<<(total + 255) / 256, 256, 0, stream>>>(W_hh, whpack, HID, KC_H);
  }
  bias_kernel<<<(4 * HID + 255) / 256, 256, 0, stream>>>(b_ih, b_hh, bias);
  {
    int nh = (int)((size_t)2 * BATCH * HPAD * 2 / 4);   // both h buffers (dwords)
    zero_kernel<<<(nh + 255) / 256, 256, 0, stream>>>((uint32_t*)hping, nh);
    int nc = BATCH * HID;
    zero_kernel<<<(nc + 255) / 256, 256, 0, stream>>>((uint32_t*)cbuf, nc);
  }

  // recurrent scan: stream order provides the inter-step barrier
  const int blocks = WAVES / 4;   // 188 blocks * 128 threads = 752 waves
  for (int s = 0; s < SEQ; ++s) {
    const h16* hi = (s & 1) ? hpong : hping;
    h16*       ho = (s & 1) ? hping : hpong;
    lstm_step_kernel<<<blocks, 128, 0, stream>>>(xh, wipack, whpack, bias,
                                                 hi, ho, cbuf, out, s);
  }
}
